// Cross_Attention_13091060318626
// MI455X (gfx1250) — matrix operand order repaired and checked by reading, not compiled, not run on hardware
//
#include <hip/hip_runtime.h>

typedef __attribute__((ext_vector_type(16))) __bf16 v16bf;
typedef __attribute__((ext_vector_type(8)))  float  v8f;
typedef __attribute__((ext_vector_type(4)))  uint32_t u32x4;
typedef __attribute__((ext_vector_type(8)))  uint32_t u32x8;

#define EPSV 1e-6f

// ---------------------------------------------------------------------------
// Workspace layout (bf16 elements unless noted):
//   fragq : [br][24 mt][4 kt][32 lane][16 e]   -> 2*49152  bf16  @ elem 0
//   fragp : [br][ 8 mt][4 kt][32 lane][16 e]   -> 2*16384  bf16  @ elem 98304
//   Y     : [pair=br*2+b][384][16384]          -> 25165824 bf16  @ elem 131072
//   QKV   : same shape                          -> 25165824 bf16 @ elem 25296896
//   stats : [pair][8 h][attn 256 | ksum 16 | vsum 16] fp32 @ byte 100925440
// ---------------------------------------------------------------------------

// CDNA5 async copy: 16 bytes global -> LDS, tracked with ASYNCcnt.
// LDS byte offset = low 32 bits of the generic (flat) shared-memory pointer
// (ISA 10.2: LDS aperture address is truncated to addr[31:0]).
__device__ __forceinline__ void async_b128(void* lds_ptr, const void* gptr) {
  uint32_t loff = (uint32_t)(uintptr_t)lds_ptr;
  uint64_t ga = (uint64_t)(uintptr_t)gptr;
  asm volatile("global_load_async_to_lds_b128 %0, %1, off"
               :: "v"(loff), "v"(ga) : "memory");
}
__device__ __forceinline__ void wait_async0() {
  asm volatile("s_wait_asynccnt 0x0" ::: "memory");
}

// Fragment K-map for 16-bit A matrix (contiguous, matches B fragments):
//   lane<16 : e0..15 -> K0..15
//   lane>=16: e0..15 -> K16..31
__global__ __launch_bounds__(256)
void k_prep_weights(const float* __restrict__ qkv1, const float* __restrict__ qkv2,
                    const float* __restrict__ proj1, const float* __restrict__ proj2,
                    __bf16* __restrict__ fragq, __bf16* __restrict__ fragp) {
  int t = blockIdx.x * 256 + threadIdx.x;           // 131072 total
  if (t < 98304) {                                  // qkv fragments
    int br = t / 49152, r = t % 49152;
    int e = r & 15, lane = (r >> 4) & 31, kt = (r >> 9) & 3, mt = r >> 11;
    int row = mt * 16 + (lane & 15);
    int k = kt * 32 + e + ((lane < 16) ? 0 : 16);
    const float* w = br ? qkv2 : qkv1;              // [384][128]
    fragq[t] = (__bf16)w[row * 128 + k];
  } else {                                          // proj fragments
    int t2 = t - 98304;
    int br = t2 / 16384, r = t2 % 16384;
    int e = r & 15, lane = (r >> 4) & 31, kt = (r >> 9) & 3, mt = r >> 11;
    int row = mt * 16 + (lane & 15);
    int k = kt * 32 + e + ((lane < 16) ? 0 : 16);
    const float* w = br ? proj2 : proj1;            // [128][128]
    fragp[t2] = (__bf16)w[row * 128 + k];
  }
}

// QKV 1x1 conv as GEMM: Y[384 x 16384] = Wqkv[384x128] * X[128x16384], per pair.
// X tile staged via async-to-LDS DMA (fp32, verbatim), then converted in LDS
// to the pixel-major bf16 B-fragment layout.
__global__ __launch_bounds__(256)
void k_qkv_gemm(const float* __restrict__ feat, const __bf16* __restrict__ fragq,
                __bf16* __restrict__ Y) {
  int pair = blockIdx.y, br = pair >> 1, b = pair & 1;
  int p0 = blockIdx.x * 64;
  __shared__ alignas(16) float  xf[128 * 64];       // [ch][px] fp32 (async dest)
  __shared__ alignas(16) __bf16 xs[64 * 144];       // [px][ch] bf16, rowstride 144
  const float* xg = feat + ((size_t)b * 256 + (size_t)br * 128) * 16384 + p0;
  int tid = threadIdx.x;
#pragma unroll
  for (int i = 0; i < 8; ++i) {                     // 2048 granules of 16B
    int g = i * 256 + tid;
    int ch = g >> 4, gx = g & 15;                   // 16 granules per 64-px row
    async_b128(&xf[ch * 64 + gx * 4], xg + (size_t)ch * 16384 + gx * 4);
  }
  wait_async0();
  __syncthreads();
#pragma unroll
  for (int i = 0; i < 32; ++i) {                    // f32 -> bf16, transpose
    int e = i * 256 + tid;
    int ch = e >> 6, px = e & 63;
    xs[px * 144 + ch] = (__bf16)xf[ch * 64 + px];
  }
  __syncthreads();

  int wave = tid >> 5, lane = tid & 31;
  int nbase = lane & 15;
  int koffB = (lane < 16) ? 0 : 16;                 // dense-B: contiguous K per half-wave
  const __bf16* fb = fragq + br * 49152;
  v8f zero8 = {0.f, 0.f, 0.f, 0.f, 0.f, 0.f, 0.f, 0.f};
  v8f acc[3][4];
#pragma unroll
  for (int i = 0; i < 3; ++i)
#pragma unroll
    for (int j = 0; j < 4; ++j) acc[i][j] = zero8;

#pragma unroll
  for (int kt = 0; kt < 4; ++kt) {
    v16bf bfr[4];
#pragma unroll
    for (int nt = 0; nt < 4; ++nt)
      bfr[nt] = *(const v16bf*)&xs[(nt * 16 + nbase) * 144 + kt * 32 + koffB];
#pragma unroll
    for (int i = 0; i < 3; ++i) {
      int mt = wave * 3 + i;
      v16bf afr = *(const v16bf*)(fb + ((size_t)(mt * 4 + kt) * 32 + lane) * 16);
#pragma unroll
      for (int nt = 0; nt < 4; ++nt)
        acc[i][nt] = __builtin_amdgcn_wmma_f32_16x16x32_bf16(
            false, afr, false, bfr[nt], (short)0, acc[i][nt], false, false);
    }
  }

  __bf16* yout = Y + (size_t)pair * 384 * 16384 + p0;
  int mrow = (lane < 16) ? 0 : 8;
#pragma unroll
  for (int i = 0; i < 3; ++i) {
    int mt = wave * 3 + i;
#pragma unroll
    for (int nt = 0; nt < 4; ++nt) {
      int p = nt * 16 + nbase;
#pragma unroll
      for (int r = 0; r < 8; ++r)
        yout[(size_t)(mt * 16 + r + mrow) * 16384 + p] = (__bf16)acc[i][nt][r];
    }
  }
}

// Depthwise 3x3, zero padding. 4 pixels per thread.
__global__ __launch_bounds__(256)
void k_dwconv(const __bf16* __restrict__ Y, const float* __restrict__ dw1,
              const float* __restrict__ dw2, __bf16* __restrict__ QKV) {
  int t = blockIdx.x * 256 + threadIdx.x;           // 6,291,456 total
  int xc = t & 31, y = (t >> 5) & 127, pc = t >> 12;  // pc = pair*384+ch in [0,1536)
  int pair = pc / 384, ch = pc % 384;
  int br = pair >> 1;
  const float* w = (br ? dw2 : dw1) + ch * 9;
  const __bf16* in = Y + (size_t)pc * 16384;
  float wv[9];
#pragma unroll
  for (int i = 0; i < 9; ++i) wv[i] = w[i];
  int x0 = xc * 4;
  float acc[4] = {0.f, 0.f, 0.f, 0.f};
#pragma unroll
  for (int ky = 0; ky < 3; ++ky) {
    int yy = y + ky - 1;
    if (yy < 0 || yy > 127) continue;
#pragma unroll
    for (int kx = 0; kx < 3; ++kx) {
      float wk = wv[ky * 3 + kx];
#pragma unroll
      for (int j = 0; j < 4; ++j) {
        int xx = x0 + j + kx - 1;
        if (xx >= 0 && xx <= 127) acc[j] += wk * (float)in[yy * 128 + xx];
      }
    }
  }
  __bf16* out = QKV + (size_t)pc * 16384 + y * 128 + x0;
#pragma unroll
  for (int j = 0; j < 4; ++j) out[j] = (__bf16)acc[j];
}

__global__ __launch_bounds__(256)
void k_zero_stats(float* __restrict__ stats) {
  int t = blockIdx.x * 256 + threadIdx.x;
  if (t < 9216) stats[t] = 0.f;
}

// Per (pair, head): attn[16][16] += k_norm * v^T over a 1024-pixel chunk (WMMA).
// The k tile (16 rows x 2048B, row stride 32KB) is fetched by the Tensor Data
// Mover: wave 0 builds the D# (group0 + group1) and issues tensor_load_to_lds,
// tracked with TENSORcnt.
__global__ __launch_bounds__(256)
void k_stats(const __bf16* __restrict__ QKV, float* __restrict__ stats) {
  int combo = blockIdx.y, pair = combo >> 3, h = combo & 7;
  int p0 = blockIdx.x * 1024;
  const __bf16* kg = QKV + ((size_t)pair * 384 + 128 + h * 16) * 16384 + p0;
  const __bf16* vg = QKV + ((size_t)pair * 384 + 256 + h * 16) * 16384 + p0;
  float* st = stats + (size_t)combo * 288;
  __shared__ alignas(16) __bf16 ks[16 * 1024];      // [c][px], rowstride 1024
  __shared__ float aux[32];
  int tid = threadIdx.x;
  if (tid < 32) aux[tid] = 0.f;
  if (tid < 32) {                                   // wave 0 issues the TDM load
    uint64_t ga = (uint64_t)(uintptr_t)kg;
    uint32_t la = (uint32_t)(uintptr_t)&ks[0];
    u32x4 g0;
    g0[0] = 1u;                                     // count=1, user mode
    g0[1] = la;                                     // lds_addr
    g0[2] = (uint32_t)ga;                           // global_addr[31:0]
    g0[3] = (uint32_t)((ga >> 32) & 0x01FFFFFFu) | (2u << 30);  // addr[56:32]|type=2
    u32x8 g1;
    g1[0] = (1u << 16);                             // data_size=1 -> 2 bytes
    g1[1] = (16384u & 0xFFFFu) << 16;               // tensor_dim0 lo16 (=16384)
    g1[2] = (16u) << 16;                            // dim0 hi=0 | tensor_dim1 lo=16
    g1[3] = (1024u) << 16;                          // dim1 hi=0 | tile_dim0=1024
    g1[4] = 16u;                                    // tile_dim1=16
    g1[5] = 16384u;                                 // tensor_dim0_stride lo32
    g1[6] = 0u;
    g1[7] = 0u;
    asm volatile("tensor_load_to_lds %0, %1" :: "s"(g0), "s"(g1) : "memory");
    __builtin_amdgcn_s_wait_tensorcnt((short)0);
  }
  __syncthreads();

  // normalize per pixel in LDS; accumulate ksum partials
  float ksump[16];
#pragma unroll
  for (int c = 0; c < 16; ++c) ksump[c] = 0.f;
  for (int i = 0; i < 4; ++i) {
    int px = i * 256 + tid;
    float kv[16], ss = 0.f;
#pragma unroll
    for (int c = 0; c < 16; ++c) { kv[c] = (float)ks[c * 1024 + px]; ss += kv[c] * kv[c]; }
    float inv = 1.f / (sqrtf(ss) + EPSV);
#pragma unroll
    for (int c = 0; c < 16; ++c) {
      float kn = kv[c] * inv;
      ks[c * 1024 + px] = (__bf16)kn;
      ksump[c] += kn;
    }
  }
  // vsum partials: thread handles d = tid&15 over 64 pixels
  float vsump = 0.f;
  {
    int d = tid & 15, g = tid >> 4;
    const __bf16* vr = vg + (size_t)d * 16384 + g * 64;
    for (int j = 0; j < 64; ++j) vsump += (float)vr[j];
  }
#pragma unroll
  for (int c = 0; c < 16; ++c) atomicAdd(&aux[c], ksump[c]);
  atomicAdd(&aux[16 + (tid & 15)], vsump);
  __syncthreads();
  if (tid < 16) {
    atomicAdd(&st[256 + tid], aux[tid]);
    atomicAdd(&st[272 + tid], aux[16 + tid]);
  }

  // attn via WMMA: A = k_norm (LDS), B = v (global, layout already matches B frag)
  int wave = tid >> 5, lane = tid & 31;
  int nbase = lane & 15;
  int koffA = (lane < 16) ? 0 : 16;
  int koffB = (lane < 16) ? 0 : 16;
  v8f acc = {0.f, 0.f, 0.f, 0.f, 0.f, 0.f, 0.f, 0.f};
#pragma unroll
  for (int s = 0; s < 4; ++s) {
    int pb = (wave * 4 + s) * 32;
    const __bf16* pa = &ks[nbase * 1024 + pb + koffA];
    v16bf afr;
#pragma unroll
    for (int e = 0; e < 8; ++e) { afr[e] = pa[e]; afr[8 + e] = pa[8 + e]; }
    v16bf bfr = *(const v16bf*)(vg + (size_t)nbase * 16384 + pb + koffB);
    acc = __builtin_amdgcn_wmma_f32_16x16x32_bf16(false, afr, false, bfr,
                                                  (short)0, acc, false, false);
  }
  int mrow = (lane < 16) ? 0 : 8;
#pragma unroll
  for (int r = 0; r < 8; ++r)
    atomicAdd(&st[(r + mrow) * 16 + nbase], acc[r]);
}

// Branch output + proj 1x1 conv (WMMA), fused per 64-pixel tile.
// q tile staged via async-to-LDS DMA (bf16, verbatim).
__global__ __launch_bounds__(256)
void k_out(const __bf16* __restrict__ QKV, const float* __restrict__ stats,
           const __bf16* __restrict__ fragp, const float* __restrict__ temp1,
           const float* __restrict__ temp2, float* __restrict__ out) {
  int pair = blockIdx.y, br = pair >> 1, b = pair & 1;
  int p0 = blockIdx.x * 64;
  __shared__ alignas(16) __bf16 qs[128 * 72];       // raw q, [ch][px], rowstride 72
  __shared__ alignas(16) __bf16 os[64 * 144];       // pre-proj out, [px][ch]
  __shared__ float sinv[8 * 64], sden[8 * 64];
  __shared__ float sst[8 * 288];                    // other-branch stats
  __shared__ float stemp[8];
  int tid = threadIdx.x;
  const __bf16* qg = QKV + (size_t)pair * 384 * 16384 + p0;  // q = channels 0..127
#pragma unroll
  for (int i = 0; i < 4; ++i) {                     // 1024 granules of 16B (8 bf16)
    int g = i * 256 + tid;
    int ch = g >> 3, gx = g & 7;
    async_b128(&qs[ch * 72 + gx * 8], qg + (size_t)ch * 16384 + gx * 8);
  }
  {
    const float* so = stats + (size_t)((1 - br) * 2 + b) * 8 * 288;
    for (int e = tid; e < 2304; e += 256) sst[e] = so[e];
    if (tid < 8) stemp[tid] = (br ? temp2 : temp1)[tid];
  }
  wait_async0();
  __syncthreads();

  // per (head, px): q-norm scale + denominator
#pragma unroll
  for (int i = 0; i < 2; ++i) {
    int e = i * 256 + tid;                          // 512 = 8 heads * 64 px
    int hh = e >> 6, px = e & 63;
    const __bf16* qcol = &qs[(hh * 16) * 72 + px];
    const float* s = &sst[hh * 288];
    float ss = 0.f, dk = 0.f;
#pragma unroll
    for (int c = 0; c < 16; ++c) {
      float qv = (float)qcol[c * 72];
      ss += qv * qv;
      dk += qv * s[256 + c];
    }
    float inv = 1.f / (sqrtf(ss) + EPSV);
    sinv[e] = inv;
    sden[e] = 16384.f + dk * inv + EPSV;
  }
  __syncthreads();

  // pre-projection output, stored pixel-major for B fragments
#pragma unroll
  for (int i = 0; i < 32; ++i) {
    int e = i * 256 + tid;
    int ch = e >> 6, px = e & 63;
    int hh = ch >> 4, d = ch & 15;
    const float* s = &sst[hh * 288];
    const __bf16* qcol = &qs[(hh * 16) * 72 + px];
    float a = 0.f;
#pragma unroll
    for (int c = 0; c < 16; ++c) a += (float)qcol[c * 72] * s[c * 16 + d];
    float inv = sinv[hh * 64 + px], den = sden[hh * 64 + px];
    os[px * 144 + ch] = (__bf16)(stemp[hh] * (s[272 + d] + a * inv) / den);
  }
  __syncthreads();

  // proj GEMM: D[128 x 64] = Wproj[128x128] * OS[128x64]
  int wave = tid >> 5, lane = tid & 31;
  int nbase = lane & 15;
  int koffB = (lane < 16) ? 0 : 16;
  int mrow = (lane < 16) ? 0 : 8;
  const __bf16* fb = fragp + br * 16384;
  v8f zero8 = {0.f, 0.f, 0.f, 0.f, 0.f, 0.f, 0.f, 0.f};
  v8f acc[4] = {zero8, zero8, zero8, zero8};
#pragma unroll
  for (int kt = 0; kt < 4; ++kt) {
    v16bf afr = *(const v16bf*)(fb + ((size_t)(wave * 4 + kt) * 32 + lane) * 16);
#pragma unroll
    for (int nt = 0; nt < 4; ++nt) {
      v16bf bfr = *(const v16bf*)&os[(nt * 16 + nbase) * 144 + kt * 32 + koffB];
      acc[nt] = __builtin_amdgcn_wmma_f32_16x16x32_bf16(
          false, afr, false, bfr, (short)0, acc[nt], false, false);
    }
  }
  float* og = out + ((size_t)b * 256 + (size_t)br * 128) * 16384 + p0;
#pragma unroll
  for (int nt = 0; nt < 4; ++nt) {
    int p = nt * 16 + nbase;
#pragma unroll
    for (int r = 0; r < 8; ++r)
      og[(size_t)(wave * 16 + r + mrow) * 16384 + p] = acc[nt][r];
  }
}

extern "C" void kernel_launch(void* const* d_in, const int* in_sizes, int n_in,
                              void* d_out, int out_size, void* d_ws, size_t ws_size,
                              hipStream_t stream) {
  const float* feat  = (const float*)d_in[0];
  const float* qkv1  = (const float*)d_in[1];
  const float* dw1   = (const float*)d_in[2];
  const float* proj1 = (const float*)d_in[3];
  const float* qkv2  = (const float*)d_in[4];
  const float* dw2   = (const float*)d_in[5];
  const float* proj2 = (const float*)d_in[6];
  const float* temp1 = (const float*)d_in[7];
  const float* temp2 = (const float*)d_in[8];
  float* out = (float*)d_out;

  __bf16* wsb   = (__bf16*)d_ws;
  __bf16* fragq = wsb;                         // 98304 bf16
  __bf16* fragp = wsb + 98304;                 // 32768 bf16
  __bf16* Y     = wsb + 131072;                // 25165824 bf16
  __bf16* QKV   = wsb + 131072 + 25165824;     // 25165824 bf16
  float*  stats = (float*)((char*)d_ws + 100925440);  // 9216 fp32

  k_prep_weights<<<512, 256, 0, stream>>>(qkv1, qkv2, proj1, proj2, fragq, fragp);
  k_qkv_gemm<<<dim3(256, 4), 256, 0, stream>>>(feat, fragq, Y);
  k_dwconv<<<24576, 256, 0, stream>>>(Y, dw1, dw2, QKV);
  k_zero_stats<<<36, 256, 0, stream>>>(stats);
  k_stats<<<dim3(16, 32), 256, 0, stream>>>(QKV, stats);
  k_out<<<dim3(256, 4), 256, 0, stream>>>(QKV, stats, fragp, temp1, temp2, out);
}
